// MultiSpanAllocator_6614249636435
// MI455X (gfx1250) — matrix operand order repaired and checked by reading, not compiled, not run on hardware
//
#include <hip/hip_runtime.h>

typedef __attribute__((ext_vector_type(16))) _Float16 v16h;
typedef __attribute__((ext_vector_type(8)))  _Float16 v8h;
typedef __attribute__((ext_vector_type(4)))  _Float16 v4h;
typedef __attribute__((ext_vector_type(8)))  float    v8f;

#define NN      4096
#define HH      8
#define DD      64
#define BM      64      // q rows per block
#define BN      64      // keys per iteration
#define WM      16      // q rows per wave
#define RADIUS2 6.25f
#define NEGVAL  (-1e30f)
#define SCALE   0.125f  // 1/sqrt(64)

union V16U { v16h v; v8h h[2]; };

// Rotate-butterfly reductions over each 16-lane row via DPP16 row_ror
// (VALU-only: no ds_bpermute, no DScnt waits).
__device__ __forceinline__ float red_max16(float x) {
    int t;
    t = __builtin_amdgcn_update_dpp(0, __float_as_int(x), 0x121, 0xf, 0xf, true);
    x = fmaxf(x, __int_as_float(t));
    t = __builtin_amdgcn_update_dpp(0, __float_as_int(x), 0x122, 0xf, 0xf, true);
    x = fmaxf(x, __int_as_float(t));
    t = __builtin_amdgcn_update_dpp(0, __float_as_int(x), 0x124, 0xf, 0xf, true);
    x = fmaxf(x, __int_as_float(t));
    t = __builtin_amdgcn_update_dpp(0, __float_as_int(x), 0x128, 0xf, 0xf, true);
    x = fmaxf(x, __int_as_float(t));
    return x;
}
__device__ __forceinline__ float red_sum16(float x) {
    int t;
    t = __builtin_amdgcn_update_dpp(0, __float_as_int(x), 0x121, 0xf, 0xf, true);
    x += __int_as_float(t);
    t = __builtin_amdgcn_update_dpp(0, __float_as_int(x), 0x122, 0xf, 0xf, true);
    x += __int_as_float(t);
    t = __builtin_amdgcn_update_dpp(0, __float_as_int(x), 0x124, 0xf, 0xf, true);
    x += __int_as_float(t);
    t = __builtin_amdgcn_update_dpp(0, __float_as_int(x), 0x128, 0xf, 0xf, true);
    x += __int_as_float(t);
    return x;
}

// ---------------------------------------------------------------------------
// Prep: K -> f16 row-major [h][n][d];  V -> f16 transposed [h][d][n]
// ---------------------------------------------------------------------------
__global__ __launch_bounds__(256)
void msa_preconvert(const float* __restrict__ k, const float* __restrict__ v,
                    _Float16* __restrict__ kf, _Float16* __restrict__ vt)
{
    const int h = blockIdx.y;
    const int e = (blockIdx.x * 256 + threadIdx.x) * 4;   // over N*D floats
    const int n = e >> 6;                                 // DD = 64
    const int d = e & 63;
    const size_t base = (size_t)h * NN * DD;

    const float4 k4 = *(const float4*)&k[base + e];
    const float4 v4 = *(const float4*)&v[base + e];

    v4h kp;
    kp[0] = (_Float16)k4.x; kp[1] = (_Float16)k4.y;
    kp[2] = (_Float16)k4.z; kp[3] = (_Float16)k4.w;
    *(v4h*)&kf[base + e] = kp;

    _Float16* vtb = vt + (size_t)h * DD * NN;
    vtb[(size_t)(d + 0) * NN + n] = (_Float16)v4.x;
    vtb[(size_t)(d + 1) * NN + n] = (_Float16)v4.y;
    vtb[(size_t)(d + 2) * NN + n] = (_Float16)v4.z;
    vtb[(size_t)(d + 3) * NN + n] = (_Float16)v4.w;
}

// ---------------------------------------------------------------------------
// Flash attention. PRE=true: double-buffered async-to-LDS pipeline.
// ---------------------------------------------------------------------------
template<bool PRE>
__global__ __launch_bounds__(128)
void msa_flash_attn(const float* __restrict__ q,
                    const float* __restrict__ kf32,
                    const float* __restrict__ vf32,
                    const _Float16* __restrict__ kf16,   // [H][N][D]
                    const _Float16* __restrict__ vf16t,  // [H][D][N]
                    const float* __restrict__ coeffs,
                    const int* __restrict__ span_ids,
                    const unsigned char* __restrict__ is_causal,
                    float* __restrict__ out)
{
    __shared__ _Float16 Ksh[2][BN * DD];   // [buf][key][d]  row-major
    __shared__ _Float16 Vsh[2][DD * BN];   // [buf][d][key]  transposed
    __shared__ _Float16 Psh[4][WM * BN];   // per-wave P staging [row][key]

    const int tid  = threadIdx.x;
    const int lane = tid & 31;
    const int wave = tid >> 5;
    const int ln16 = lane & 15;
    const int hl   = lane >> 4;            // 0 or 1 (16-lane half)

    const int head = blockIdx.y;
    const int bq0  = blockIdx.x * BM;
    const int q0   = bq0 + wave * WM;      // first q row of this wave

    const float* qh = q + (size_t)head * NN * DD;
    float*       oh = out + (size_t)head * NN * DD;

    // scalar span bounds (span ids are monotonic contiguous segments)
    const int bqs_hi = span_ids[bq0 + BM - 1];   // block-level max q span
    const int wqs_lo = span_ids[q0];             // wave-level min q span

    // Future-span tiles (fully masked, exactly zero probability) form a
    // suffix under monotonic spans: binary-search the loop end (6 loads).
    int lo = 0, hi = NN / BN;
    while (lo < hi) {
        const int mid = (lo + hi) >> 1;
        if (span_ids[mid * BN] > bqs_hi) hi = mid; else lo = mid + 1;
    }
    const int kend = lo * BN;              // >= BN (own span never skipped)

    // ---- Q A-fragments (rows q0..q0+15, d 0..63), f32 -> f16 ----
    v16h qa[2];
    {
        const float* qrow = qh + (size_t)(q0 + ln16) * DD;
        #pragma unroll
        for (int f = 0; f < 2; ++f) {
            #pragma unroll
            for (int i = 0; i < 8; ++i) {
                qa[f][i]     = (_Float16)qrow[32*f + 16*hl +     i];
                qa[f][8 + i] = (_Float16)qrow[32*f + 16*hl + 8 + i];
            }
        }
    }

    // ---- per-row q metadata (C-layout rows: r = q0 + vv + 8*hl) ----
    int   qspan[8], qcau[8], qix[8];
    float qcx[8], qcy[8];
    #pragma unroll
    for (int vv = 0; vv < 8; ++vv) {
        const int r = q0 + vv + 8*hl;
        qix[vv]   = r;
        qspan[vv] = span_ids[r];
        qcau[vv]  = (int)is_causal[r];
        qcx[vv]   = coeffs[2*r + 0];
        qcy[vv]   = coeffs[2*r + 1];
    }

    // ---- accumulators ----
    v8f Oa[4];
    float mrow[8], lrow[8];
    #pragma unroll
    for (int t = 0; t < 4; ++t)
        #pragma unroll
        for (int i = 0; i < 8; ++i) Oa[t][i] = 0.0f;
    #pragma unroll
    for (int i = 0; i < 8; ++i) { mrow[i] = -INFINITY; lrow[i] = 0.0f; }

    const float log2e = 1.44269504088896340736f;
    _Float16* pp = &Psh[wave][0];

    // async-copy one 64-key K/V f16 tile into LDS buffer b (ASYNCcnt)
    auto issue_tile = [&](int k0, int b) {
        const _Float16* ksrc = kf16 + ((size_t)head * NN + k0) * DD;
        const _Float16* vsrc = vf16t + (size_t)head * DD * NN + k0;
        #pragma unroll
        for (int i = 0; i < 4; ++i) {
            const int c = i * 128 + tid;                 // 0..511 b128 chunks
            unsigned kl = (unsigned)(size_t)&Ksh[b][c * 8];
            unsigned long long kg = (unsigned long long)(size_t)(ksrc + c * 8);
            asm volatile("global_load_async_to_lds_b128 %0, %1, off"
                         :: "v"(kl), "v"(kg) : "memory");
            const int dr = c >> 3, pt = c & 7;
            unsigned vl = (unsigned)(size_t)&Vsh[b][dr * BN + pt * 8];
            unsigned long long vg =
                (unsigned long long)(size_t)(vsrc + (size_t)dr * NN + pt * 8);
            asm volatile("global_load_async_to_lds_b128 %0, %1, off"
                         :: "v"(vl), "v"(vg) : "memory");
        }
    };

    int buf = 0;
    if constexpr (PRE) issue_tile(0, 0);   // prologue: tile 0 never skipped

    for (int k0 = 0; k0 < kend; k0 += BN) {
        const _Float16 *Kb, *Vb;
        if constexpr (PRE) {
            asm volatile("s_wait_asynccnt 0x0" ::: "memory");
            __syncthreads();   // all copies landed; all waves done with other buf
            if (k0 + BN < kend) issue_tile(k0 + BN, buf ^ 1);
            Kb = Ksh[buf];
            Vb = Vsh[buf];
            buf ^= 1;
        } else {
            __syncthreads();
            const float* kh = kf32 + (size_t)head * NN * DD;
            const float* vh = vf32 + (size_t)head * NN * DD;
            #pragma unroll
            for (int i = 0; i < 8; ++i) {
                const int e  = (i * 128 + tid) * 4;
                const int kr = e >> 6;
                const int kc = e & 63;
                const float4 k4 = *(const float4*)&kh[(size_t)(k0 + kr) * DD + kc];
                const float4 v4 = *(const float4*)&vh[(size_t)(k0 + kr) * DD + kc];
                v4h kp;
                kp[0] = (_Float16)k4.x; kp[1] = (_Float16)k4.y;
                kp[2] = (_Float16)k4.z; kp[3] = (_Float16)k4.w;
                *(v4h*)&Ksh[0][kr * DD + kc] = kp;
                Vsh[0][(kc + 0) * BN + kr] = (_Float16)v4.x;
                Vsh[0][(kc + 1) * BN + kr] = (_Float16)v4.y;
                Vsh[0][(kc + 2) * BN + kr] = (_Float16)v4.z;
                Vsh[0][(kc + 3) * BN + kr] = (_Float16)v4.w;
            }
            __syncthreads();
            Kb = Ksh[0];
            Vb = Vsh[0];
        }

        // ---- S = Q * K^T : 4 n-tiles x (D=64 -> 2 WMMAs) ----
        v8f S[4];
        #pragma unroll
        for (int nt = 0; nt < 4; ++nt) {
            v8f acc = {0.f,0.f,0.f,0.f,0.f,0.f,0.f,0.f};
            const v16h b0 = *(const v16h*)&Kb[(nt*16 + ln16) * DD +      16*hl];
            const v16h b1 = *(const v16h*)&Kb[(nt*16 + ln16) * DD + 32 + 16*hl];
            acc = __builtin_amdgcn_wmma_f32_16x16x32_f16(false, qa[0], false, b0, (short)0, acc, false, false);
            acc = __builtin_amdgcn_wmma_f32_16x16x32_f16(false, qa[1], false, b1, (short)0, acc, false, false);
            S[nt] = acc;
        }

        // ---- scale + mask ----
        const int ks_hi = span_ids[k0 + BN - 1];
        if (wqs_lo > ks_hi) {
            // pure history tile for this wave: mask is all-true
            #pragma unroll
            for (int nt = 0; nt < 4; ++nt)
                #pragma unroll
                for (int vv = 0; vv < 8; ++vv)
                    S[nt][vv] *= SCALE;
        } else {
            #pragma unroll
            for (int nt = 0; nt < 4; ++nt) {
                const int   kidx  = k0 + nt*16 + ln16;
                const int   kspan = span_ids[kidx];
                const float kcx   = coeffs[2*kidx + 0];
                const float kcy   = coeffs[2*kidx + 1];
                #pragma unroll
                for (int vv = 0; vv < 8; ++vv) {
                    const float dx = qcx[vv] - kcx;
                    const float dy = qcy[vv] - kcy;
                    const float d2 = dx*dx + dy*dy;
                    const bool ok = (qspan[vv] > kspan) ||
                                    ((qspan[vv] == kspan) &&
                                     ((qcau[vv] == 0) || (qix[vv] >= kidx)) &&
                                     (d2 < RADIUS2));
                    S[nt][vv] = ok ? S[nt][vv] * SCALE : NEGVAL;
                }
            }
        }

        // ---- online softmax update (DPP row reductions, no DS traffic) ----
        float alpha[8];
        #pragma unroll
        for (int vv = 0; vv < 8; ++vv) {
            float mx = fmaxf(fmaxf(S[0][vv], S[1][vv]), fmaxf(S[2][vv], S[3][vv]));
            mx = red_max16(mx);
            const float mnew = fmaxf(mrow[vv], mx);
            alpha[vv] = __builtin_amdgcn_exp2f((mrow[vv] - mnew) * log2e);
            float psum = 0.0f;
            #pragma unroll
            for (int nt = 0; nt < 4; ++nt) {
                const float p = __builtin_amdgcn_exp2f((S[nt][vv] - mnew) * log2e);
                S[nt][vv] = p;
                psum += p;
            }
            psum = red_sum16(psum);
            lrow[vv] = lrow[vv] * alpha[vv] + psum;
            mrow[vv] = mnew;
        }
        #pragma unroll
        for (int t = 0; t < 4; ++t)
            #pragma unroll
            for (int vv = 0; vv < 8; ++vv) Oa[t][vv] *= alpha[vv];

        // ---- stage P (C-layout f32 -> row-major f16, wave-private LDS) ----
        #pragma unroll
        for (int nt = 0; nt < 4; ++nt)
            #pragma unroll
            for (int vv = 0; vv < 8; ++vv)
                pp[(vv + 8*hl) * BN + nt*16 + ln16] = (_Float16)S[nt][vv];

        // ---- O += P * V : 4 d-tiles x (keys 64 -> 2 WMMAs) ----
        #pragma unroll
        for (int t = 0; t < 4; ++t) {
            v8f acc = Oa[t];
            #pragma unroll
            for (int f = 0; f < 2; ++f) {
                V16U u;
                const _Float16* prow = pp + ln16 * BN + 32*f;
                u.h[0] = *(const v8h*)(prow + 16*hl);
                u.h[1] = *(const v8h*)(prow + 16*hl + 8);
                const v16h vb = *(const v16h*)&Vb[(t*16 + ln16) * BN + 32*f + 16*hl];
                acc = __builtin_amdgcn_wmma_f32_16x16x32_f16(false, u.v, false, vb, (short)0, acc, false, false);
            }
            Oa[t] = acc;
        }
    }

    // ---- normalize and store ----
    float inv[8];
    #pragma unroll
    for (int vv = 0; vv < 8; ++vv) inv[vv] = 1.0f / lrow[vv];
    #pragma unroll
    for (int t = 0; t < 4; ++t)
        #pragma unroll
        for (int vv = 0; vv < 8; ++vv)
            oh[(size_t)(q0 + vv + 8*hl) * DD + t*16 + ln16] = Oa[t][vv] * inv[vv];
}

extern "C" void kernel_launch(void* const* d_in, const int* in_sizes, int n_in,
                              void* d_out, int out_size, void* d_ws, size_t ws_size,
                              hipStream_t stream) {
    const float*         q         = (const float*)d_in[0];
    const float*         k         = (const float*)d_in[1];
    const float*         v         = (const float*)d_in[2];
    const float*         coeffs    = (const float*)d_in[3];
    const int*           span_ids  = (const int*)d_in[4];
    const unsigned char* is_causal = (const unsigned char*)d_in[5];
    float*               out       = (float*)d_out;

    const size_t need = (size_t)2 * HH * NN * DD * sizeof(_Float16);  // 8 MB
    dim3 grid(NN / BM, HH, 1);

    if (ws_size >= need) {
        _Float16* kf = (_Float16*)d_ws;
        _Float16* vt = kf + (size_t)HH * NN * DD;
        dim3 pgrid((NN * DD) / (256 * 4), HH, 1);
        msa_preconvert<<<pgrid, 256, 0, stream>>>(k, v, kf, vt);
        msa_flash_attn<true><<<grid, 128, 0, stream>>>(
            q, k, v, kf, vt, coeffs, span_ids, is_causal, out);
    } else {
        msa_flash_attn<false><<<grid, 128, 0, stream>>>(
            q, k, v, nullptr, nullptr, coeffs, span_ids, is_causal, out);
    }
}
